// CausalSelfAttention_56959856280149
// MI455X (gfx1250) — matrix operand order repaired and checked by reading, not compiled, not run on hardware
//
#include <hip/hip_runtime.h>
#include <hip/hip_bf16.h>

// ---------------------------------------------------------------------------
// Causal self-attention forward for MI455X (gfx1250, wave32, WMMA).
// f32->bf16 convert, 3 WMMA projection GEMMs (32x64 per-wave tile, 2-stage
// software pipeline), fused causal flash-attention (online softmax, WMMA for
// QK^T and P*V), final WMMA GEMM to f32.
// ---------------------------------------------------------------------------

typedef __attribute__((ext_vector_type(16))) __bf16 v16bf;
typedef __attribute__((ext_vector_type(8)))  __bf16 bf16x8;
typedef __attribute__((ext_vector_type(8)))  float  v8f;

constexpr int kE = 1024;          // embedding dim
constexpr int kS = 2048;          // sequence length
constexpr int kB = 2;             // batch
constexpr int kH = 16;            // heads
constexpr int kD = 64;            // head dim
constexpr int kM = kB * kS;       // flattened rows (B*S)

// Build a 16-element bf16 fragment from two contiguous 8-element (16B) loads.
__device__ __forceinline__ v16bf load_frag(const __bf16* lo, const __bf16* hi) {
  bf16x8 a = *(const bf16x8*)lo;
  bf16x8 b = *(const bf16x8*)hi;
  v16bf r;
#pragma unroll
  for (int i = 0; i < 8; ++i) { r[i] = a[i]; r[i + 8] = b[i]; }
  return r;
}

__device__ __forceinline__ v8f wmma_bf16(v16bf a, v16bf b, v8f c) {
  // (neg_a, A, neg_b, B, c_mod, C, reuse_a, reuse_b)
  return __builtin_amdgcn_wmma_f32_16x16x32_bf16(false, a, false, b,
                                                 (short)0, c, false, false);
}

// ---------------------------------------------------------------------------
// Elementwise f32 -> bf16
// ---------------------------------------------------------------------------
__global__ void cvt_f32_bf16(const float* __restrict__ in,
                             __bf16* __restrict__ out, int n) {
  int i = blockIdx.x * blockDim.x + threadIdx.x;
  if (i < n) out[i] = (__bf16)in[i];
}

// W (kE x kE row-major, indexed [k][n]) -> WT bf16 (indexed [n][k]) so that
// WMMA B-fragments read contiguous 8-element runs along k.
__global__ void cvt_w_transpose(const float* __restrict__ W,
                                __bf16* __restrict__ WT) {
  int idx = blockIdx.x * blockDim.x + threadIdx.x;
  if (idx >= kE * kE) return;
  int k = idx / kE, n = idx % kE;            // coalesced read along n
  WT[(size_t)n * kE + k] = (__bf16)W[idx];
}

// ---------------------------------------------------------------------------
// GEMM: Out(M x kE) = A(M x kE, bf16 row-major) * W (via WT = W^T, bf16).
// Block = 256 threads = 8 waves; each wave computes a 32x64 tile
// (8 accumulators). k-loop is unrolled by 2 with two fragment sets so each
// WMMA consumes fragments loaded one stage earlier (partial loadcnt waits,
// load/compute overlap, no register copies).
// mode 0: write bf16, (B,H,S,D) layout          (Q, K)
// mode 1: write bf16, (B,H,D,S) layout          (V transposed for P*V frags)
// mode 2: write f32 row-major (B*S, kE)         (final projection -> d_out)
// ---------------------------------------------------------------------------
struct Frags {
  v16bf a[2];   // two 16-row A tiles
  v16bf b[4];   // four 16-col B tiles
};

__device__ __forceinline__ Frags load_stage(const __bf16* __restrict__ A,
                                            const __bf16* __restrict__ WT,
                                            int mbase, int nbase, int k0,
                                            int l16, int half) {
  Frags f;
#pragma unroll
  for (int r = 0; r < 2; ++r) {
    const __bf16* arow = A + (size_t)(mbase + r * 16 + l16) * kE + k0 + half * 8;
    f.a[r] = load_frag(arow, arow + 16);
  }
#pragma unroll
  for (int nt = 0; nt < 4; ++nt) {
    const __bf16* bcol =
        WT + (size_t)(nbase + nt * 16 + l16) * kE + k0 + half * 8;
    f.b[nt] = load_frag(bcol, bcol + 16);
  }
  return f;
}

__global__ void __launch_bounds__(256)
gemm_bf16_wmma(const __bf16* __restrict__ A, const __bf16* __restrict__ WT,
               void* __restrict__ out, int mode) {
  const int lane = threadIdx.x & 31;
  const int wave = threadIdx.x >> 5;
  const int half = lane >> 4;      // 0: lanes 0-15, 1: lanes 16-31
  const int l16  = lane & 15;
  const int mbase = blockIdx.y * 256 + wave * 32;
  const int nbase = blockIdx.x * 64;

  v8f acc[2][4] = {};

  // 2-stage pipeline: f0 holds k0, loads for k0+32 issue before f0's WMMAs.
  Frags f0 = load_stage(A, WT, mbase, nbase, 0, l16, half);
  for (int k0 = 0; k0 < kE; k0 += 64) {
    Frags f1 = load_stage(A, WT, mbase, nbase, k0 + 32, l16, half);
#pragma unroll
    for (int nt = 0; nt < 4; ++nt)
#pragma unroll
      for (int r = 0; r < 2; ++r)
        acc[r][nt] = wmma_bf16(f0.a[r], f0.b[nt], acc[r][nt]);

    const int knext = (k0 + 64 < kE) ? k0 + 64 : 0;  // clamp (values unused)
    f0 = load_stage(A, WT, mbase, nbase, knext, l16, half);
#pragma unroll
    for (int nt = 0; nt < 4; ++nt)
#pragma unroll
      for (int r = 0; r < 2; ++r)
        acc[r][nt] = wmma_bf16(f1.a[r], f1.b[nt], acc[r][nt]);
  }

  // C/D layout: vgpr i -> row (tile base + half*8 + i), col = nt*16 + l16.
#pragma unroll
  for (int r = 0; r < 2; ++r) {
#pragma unroll
    for (int nt = 0; nt < 4; ++nt) {
#pragma unroll
      for (int i = 0; i < 8; ++i) {
        int m = mbase + r * 16 + half * 8 + i;
        int n = nbase + nt * 16 + l16;
        float v = acc[r][nt][i];
        if (mode == 2) {
          ((float*)out)[(size_t)m * kE + n] = v;
        } else {
          int b = m / kS, s = m % kS;
          int h = n / kD, d = n % kD;
          size_t idx = (mode == 0)
              ? ((size_t)(b * kH + h) * kS + s) * kD + d    // (B,H,S,D)
              : ((size_t)(b * kH + h) * kD + d) * kS + s;   // (B,H,D,S)
          ((__bf16*)out)[idx] = (__bf16)v;
        }
      }
    }
  }
}

// ---------------------------------------------------------------------------
// Fused causal flash-attention. One wave owns one 16-query tile, streams
// 32-key steps. All K and V fragment loads are hoisted to the top of the
// step (one clause) so V loads overlap the softmax VALU work; next tile is
// prefetched. P is transposed C-layout -> A-layout through padded LDS.
// ---------------------------------------------------------------------------
constexpr int kAW = 4;  // waves per block

__global__ void __launch_bounds__(kAW * 32)
attn_fwd_wmma(const __bf16* __restrict__ Qb,   // (B,H,S,D)
              const __bf16* __restrict__ Kb,   // (B,H,S,D)
              const __bf16* __restrict__ Vt,   // (B,H,D,S)
              __bf16* __restrict__ Ob) {       // (B*S, kE)
  __shared__ alignas(16) __bf16 pbuf[kAW][16][40];  // 40-col pad: conflict-free

  const int lane = threadIdx.x & 31;
  const int wave = threadIdx.x >> 5;
  const int half = lane >> 4;
  const int l16  = lane & 15;
  const int h  = blockIdx.y;
  const int b  = blockIdx.z;
  const int qs = (blockIdx.x * kAW + wave) * 16;
  const size_t bh = (size_t)b * kH + h;
  const float scale = 0.125f;  // 1/sqrt(64)

  // Q A-fragments for d in [0,32) and [32,64).
  const __bf16* qrow = Qb + (bh * kS + qs + l16) * kD;
  v16bf qf0 = load_frag(qrow +      half * 8, qrow + 16 + half * 8);
  v16bf qf1 = load_frag(qrow + 32 + half * 8, qrow + 48 + half * 8);

  v8f acc[4] = {};
  float mrow[8], lrow[8];
#pragma unroll
  for (int i = 0; i < 8; ++i) { mrow[i] = -3.0e38f; lrow[i] = 0.0f; }

  for (int ks = 0; ks < qs + 16; ks += 32) {     // causal bound on key loop
    // ---- hoisted loads: K fragments (needed now) then V (needed after
    // softmax, overlaps the VALU work below), plus next-tile prefetch.
    v16bf kf[2][2];
#pragma unroll
    for (int nt = 0; nt < 2; ++nt) {
      const __bf16* krow = Kb + (bh * kS + ks + nt * 16 + l16) * kD;
      kf[nt][0] = load_frag(krow +      half * 8, krow + 16 + half * 8);
      kf[nt][1] = load_frag(krow + 32 + half * 8, krow + 48 + half * 8);
      __builtin_prefetch(krow + 32 * kD, 0, 1);  // next key tile
    }
    v16bf vf[4];
#pragma unroll
    for (int dblk = 0; dblk < 4; ++dblk) {
      const __bf16* vp =
          Vt + (bh * kD + dblk * 16 + l16) * kS + ks + half * 8;
      vf[dblk] = load_frag(vp, vp + 16);
      __builtin_prefetch(vp + 32, 0, 1);
    }

    // ---- scores: S = Q K^T (two 16-col subtiles, K-dim = 64 = 2 WMMA each)
    float s[2][8];
#pragma unroll
    for (int nt = 0; nt < 2; ++nt) {
      v8f sc = {};
      sc = wmma_bf16(qf0, kf[nt][0], sc);
      sc = wmma_bf16(qf1, kf[nt][1], sc);
      int key = ks + nt * 16 + l16;
#pragma unroll
      for (int i = 0; i < 8; ++i) {
        int row = qs + half * 8 + i;
        float v = sc[i] * scale;
        s[nt][i] = (key <= row) ? v : -3.0e38f;  // causal mask
      }
    }

    // ---- online softmax (row stats per lane for its 8 rows)
#pragma unroll
    for (int i = 0; i < 8; ++i) {
      float rm = fmaxf(s[0][i], s[1][i]);
#pragma unroll
      for (int off = 1; off < 16; off <<= 1)     // reduce across 16 cols
        rm = fmaxf(rm, __shfl_xor(rm, off, 32));
      float mnew  = fmaxf(mrow[i], rm);
      float alpha = __expf(mrow[i] - mnew);
      float p0 = __expf(s[0][i] - mnew);
      float p1 = __expf(s[1][i] - mnew);
      float rs = p0 + p1;
#pragma unroll
      for (int off = 1; off < 16; off <<= 1)
        rs += __shfl_xor(rs, off, 32);
      lrow[i] = lrow[i] * alpha + rs;
      mrow[i] = mnew;
#pragma unroll
      for (int dblk = 0; dblk < 4; ++dblk) acc[dblk][i] *= alpha;
      // Spill P in C-layout (row = half*8+i, col = nt*16+l16).
      pbuf[wave][half * 8 + i][l16]      = (__bf16)p0;
      pbuf[wave][half * 8 + i][16 + l16] = (__bf16)p1;
    }

    __builtin_amdgcn_wave_barrier();   // order DS stores before DS loads

    // Reload P as A-fragment (row = l16; elems at half*8 and 16+half*8).
    const __bf16* prow = &pbuf[wave][l16][0];
    v16bf pf = load_frag(prow + half * 8, prow + 16 + half * 8);

    __builtin_amdgcn_wave_barrier();   // keep next iter's stores after loads

    // ---- O += P * V (V fragments already in registers)
#pragma unroll
    for (int dblk = 0; dblk < 4; ++dblk)
      acc[dblk] = wmma_bf16(pf, vf[dblk], acc[dblk]);
  }

  // Normalize and write bf16 row-major (B*S, kE) for the W_o GEMM.
#pragma unroll
  for (int dblk = 0; dblk < 4; ++dblk) {
#pragma unroll
    for (int i = 0; i < 8; ++i) {
      float o = acc[dblk][i] / lrow[i];
      size_t row = (size_t)b * kS + qs + half * 8 + i;
      Ob[row * kE + h * kD + dblk * 16 + l16] = (__bf16)o;
    }
  }
}

// ---------------------------------------------------------------------------
// Host launcher
// ---------------------------------------------------------------------------
extern "C" void kernel_launch(void* const* d_in, const int* in_sizes, int n_in,
                              void* d_out, int out_size, void* d_ws,
                              size_t ws_size, hipStream_t stream) {
  (void)in_sizes; (void)n_in; (void)out_size; (void)ws_size;

  const float* x  = (const float*)d_in[0];
  const float* Wq = (const float*)d_in[1];
  const float* Wk = (const float*)d_in[2];
  const float* Wv = (const float*)d_in[3];
  const float* Wo = (const float*)d_in[4];

  char* ws = (char*)d_ws;
  size_t off = 0;
  auto alloc = [&](size_t bytes) {
    char* p = ws + off;
    off += (bytes + 255) & ~(size_t)255;
    return p;
  };

  __bf16* xb  = (__bf16*)alloc((size_t)kM * kE * sizeof(__bf16));  // x bf16
  __bf16* wqt = (__bf16*)alloc((size_t)kE * kE * sizeof(__bf16));
  __bf16* wkt = (__bf16*)alloc((size_t)kE * kE * sizeof(__bf16));
  __bf16* wvt = (__bf16*)alloc((size_t)kE * kE * sizeof(__bf16));
  __bf16* wot = (__bf16*)alloc((size_t)kE * kE * sizeof(__bf16));
  __bf16* qb  = (__bf16*)alloc((size_t)kM * kE * sizeof(__bf16));  // (B,H,S,D)
  __bf16* kbf = (__bf16*)alloc((size_t)kM * kE * sizeof(__bf16));  // (B,H,S,D)
  __bf16* vtb = (__bf16*)alloc((size_t)kM * kE * sizeof(__bf16));  // (B,H,D,S)
  __bf16* ob  = xb;  // attention output reuses x's bf16 buffer (x dead by then)

  const int nx = kM * kE;
  cvt_f32_bf16<<<(nx + 255) / 256, 256, 0, stream>>>(x, xb, nx);

  const int nw = kE * kE;
  cvt_w_transpose<<<(nw + 255) / 256, 256, 0, stream>>>(Wq, wqt);
  cvt_w_transpose<<<(nw + 255) / 256, 256, 0, stream>>>(Wk, wkt);
  cvt_w_transpose<<<(nw + 255) / 256, 256, 0, stream>>>(Wv, wvt);
  cvt_w_transpose<<<(nw + 255) / 256, 256, 0, stream>>>(Wo, wot);

  dim3 ggrid(kE / 64, kM / 256);   // (16, 16), 256 threads, 32x64 per wave
  gemm_bf16_wmma<<<ggrid, 256, 0, stream>>>(xb, wqt, qb,  0);
  gemm_bf16_wmma<<<ggrid, 256, 0, stream>>>(xb, wkt, kbf, 0);
  gemm_bf16_wmma<<<ggrid, 256, 0, stream>>>(xb, wvt, vtb, 1);

  dim3 agrid(kS / (16 * kAW), kH, kB);  // (32, 16, 2), 128 threads
  attn_fwd_wmma<<<agrid, kAW * 32, 0, stream>>>(qb, kbf, vtb, ob);

  gemm_bf16_wmma<<<ggrid, 256, 0, stream>>>(ob, wot, d_out, 2);
}
